// UniSpeechSatGumbelVectorQuantizer_5403068859148
// MI455X (gfx1250) — matrix operand order repaired and checked by reading, not compiled, not run on hardware
//
#include <hip/hip_runtime.h>

#define HID   1024
#define NG    2
#define NV    320
#define NCOL  640        // NG*NV
#define NTOK  16384      // B*S
#define NROW  32768      // NTOK*NG
#define CVD   128        // codevector dim per group

typedef __attribute__((ext_vector_type(16))) __bf16 v16bf;
typedef __attribute__((ext_vector_type(8)))  float  v8f;
typedef __attribute__((ext_vector_type(4)))  int    v4i;

#if __has_builtin(__builtin_amdgcn_global_load_async_to_lds_b128)
#define HAS_ASYNC_LDS 1
typedef v4i __attribute__((address_space(1)))* g_v4i;  // global int4*
typedef v4i __attribute__((address_space(3)))* l_v4i;  // LDS int4*
#endif

static __device__ __forceinline__ void wait_asynccnt0() {
#if __has_builtin(__builtin_amdgcn_s_wait_asynccnt)
  __builtin_amdgcn_s_wait_asynccnt(0);
#else
  asm volatile("s_wait_asynccnt 0x0" ::: "memory");
#endif
}

static __device__ __forceinline__ void load8(const float* __restrict__ p, float* d) {
  float4 a = *(const float4*)p;          // global_load_b128
  float4 b = *(const float4*)(p + 4);    // global_load_b128
  d[0] = a.x; d[1] = a.y; d[2] = a.z; d[3] = a.w;
  d[4] = b.x; d[5] = b.y; d[6] = b.z; d[7] = b.w;
}

// ---------------------------------------------------------------------------
// Kernel 0b: one-shot split of W into bf16 hi + bf16 lo (residual) planes.
// ---------------------------------------------------------------------------
__global__ __launch_bounds__(256)
void convert_w_kernel(const float* __restrict__ W,
                      __bf16* __restrict__ whi, __bf16* __restrict__ wlo) {
  const int i = blockIdx.x * 256 + threadIdx.x;
  if (i < NCOL * HID) {
    const float x = W[i];
    const __bf16 h = (__bf16)x;
    whi[i] = h;
    wlo[i] = (__bf16)(x - (float)h);
  }
}

// ---------------------------------------------------------------------------
// Kernel 1: logits[M=16384, N=640] = H[M,1024] x W[N,1024]^T + b
// Block = 256 threads (8 waves). Block tile 64(M) x 128(N); wave grid 4Mx2N,
// each wave owns FOUR 16x16 C tiles -> 12 v_wmma per K-step (K step 32).
// Split-bf16 fp32 emulation: acc += Ahi*Bhi + Ahi*Blo + Alo*Bhi.
// B tiles (shared by all 4 M-waves) are staged in LDS via double-buffered
// global_load_async_to_lds_b128 (ASYNCcnt) and read back as ds_load_b128;
// A fragments are per-wave-unique, loaded as global b128 + v_cvt_pk_bf16_f32.
// ---------------------------------------------------------------------------
__global__ __launch_bounds__(256)
void gemm_logits_kernel(const float* __restrict__ A,
                        const __bf16* __restrict__ whi,
                        const __bf16* __restrict__ wlo,
                        const float* __restrict__ bias,
                        float* __restrict__ C) {
  const int tid   = threadIdx.x;
  const int lane  = tid & 31;
  const int wave  = tid >> 5;
  const int waveM = wave & 3;
  const int waveN = wave >> 2;
  const int mBase = blockIdx.y * 64 + waveM * 16;
  const int nBlk  = blockIdx.x * 128;            // block's first W row
  const int nBase = nBlk + waveN * 64;
  const int lr    = lane & 15;
  const int hsel  = lane >> 4;

  v8f acc[4] = {{}, {}, {}, {}};

  const float* aRow = A + (size_t)(mBase + lr) * HID + hsel * 16;

#ifdef HAS_ASYNC_LDS
  // [buf][plane(hi/lo)][row 0..127][k 0..31] bf16
  __shared__ __bf16 sW[2][2][128 * 32];

  // Per thread: row r = tid/2, 16-element half c = (tid&1)*16; 4x b128 async.
  const int cr = tid >> 1;
  const int cc = (tid & 1) * 16;
  const __bf16* gHi = whi + (size_t)(nBlk + cr) * HID + cc;
  const __bf16* gLo = wlo + (size_t)(nBlk + cr) * HID + cc;
  __bf16* dHi = &sW[0][0][cr * 32 + cc];
  __bf16* dLo = &sW[0][1][cr * 32 + cc];
  const int bufStride = 2 * 128 * 32;            // elements between buffers

#define ISSUE_WTILE(bufsel, kk)                                                      \
  do {                                                                               \
    __builtin_amdgcn_global_load_async_to_lds_b128(                                  \
        (g_v4i)(gHi + (kk)),     (l_v4i)(dHi + (bufsel) * bufStride),     0, 0);     \
    __builtin_amdgcn_global_load_async_to_lds_b128(                                  \
        (g_v4i)(gHi + (kk) + 8), (l_v4i)(dHi + (bufsel) * bufStride + 8), 0, 0);     \
    __builtin_amdgcn_global_load_async_to_lds_b128(                                  \
        (g_v4i)(gLo + (kk)),     (l_v4i)(dLo + (bufsel) * bufStride),     0, 0);     \
    __builtin_amdgcn_global_load_async_to_lds_b128(                                  \
        (g_v4i)(gLo + (kk) + 8), (l_v4i)(dLo + (bufsel) * bufStride + 8), 0, 0);     \
  } while (0)

  ISSUE_WTILE(0, 0);

  for (int k0 = 0; k0 < HID; k0 += 32) {
    const int cur = (k0 >> 5) & 1;

    wait_asynccnt0();       // this wave's copies into buf[cur] have landed
    __syncthreads();        // everyone's copies landed; prev reads of nxt done

    if (k0 + 32 < HID) ISSUE_WTILE(cur ^ 1, k0 + 32);

    float af[16];
    load8(aRow + k0,      af);
    load8(aRow + k0 + 8,  af + 8);

    v16bf Ahi, Alo;
#pragma unroll
    for (int i = 0; i < 16; ++i) {
      const __bf16 h = (__bf16)af[i];
      Ahi[i] = h;
      Alo[i] = (__bf16)(af[i] - (float)h);
    }

#pragma unroll
    for (int t = 0; t < 4; ++t) {
      const int rowT = waveN * 64 + t * 16 + lr;               // row in block tile
      const v16bf Bhi = *(const v16bf*)&sW[cur][0][rowT * 32 + hsel * 16]; // ds_load_b128 x2
      const v16bf Blo = *(const v16bf*)&sW[cur][1][rowT * 32 + hsel * 16];
      acc[t] = __builtin_amdgcn_wmma_f32_16x16x32_bf16(false, Ahi, false, Bhi, (short)0, acc[t], false, false);
      acc[t] = __builtin_amdgcn_wmma_f32_16x16x32_bf16(false, Ahi, false, Blo, (short)0, acc[t], false, false);
      acc[t] = __builtin_amdgcn_wmma_f32_16x16x32_bf16(false, Alo, false, Bhi, (short)0, acc[t], false, false);
    }
  }
#undef ISSUE_WTILE
#else
  // Fallback: direct global fragment loads (no async-to-LDS builtin available)
  const __bf16* wh[4];
  const __bf16* wl[4];
#pragma unroll
  for (int t = 0; t < 4; ++t) {
    const size_t off = (size_t)(nBase + t * 16 + lr) * HID + hsel * 16;
    wh[t] = whi + off;
    wl[t] = wlo + off;
  }
  for (int k0 = 0; k0 < HID; k0 += 32) {
    float af[16];
    load8(aRow + k0,      af);
    load8(aRow + k0 + 8,  af + 8);
    v16bf Ahi, Alo;
#pragma unroll
    for (int i = 0; i < 16; ++i) {
      const __bf16 h = (__bf16)af[i];
      Ahi[i] = h;
      Alo[i] = (__bf16)(af[i] - (float)h);
    }
#pragma unroll
    for (int t = 0; t < 4; ++t) {
      const v16bf Bhi = *(const v16bf*)(wh[t] + k0);
      const v16bf Blo = *(const v16bf*)(wl[t] + k0);
      acc[t] = __builtin_amdgcn_wmma_f32_16x16x32_bf16(false, Ahi, false, Bhi, (short)0, acc[t], false, false);
      acc[t] = __builtin_amdgcn_wmma_f32_16x16x32_bf16(false, Ahi, false, Blo, (short)0, acc[t], false, false);
      acc[t] = __builtin_amdgcn_wmma_f32_16x16x32_bf16(false, Alo, false, Bhi, (short)0, acc[t], false, false);
    }
  }
#endif

  // C 16x16 f32 layout: lanes 0-15 -> N=lane, M=vgpr; lanes 16-31 -> M=vgpr+8.
#pragma unroll
  for (int t = 0; t < 4; ++t) {
    const int col = nBase + t * 16 + lr;
    const float bi = bias[col];
#pragma unroll
    for (int e = 0; e < 8; ++e) {
      const int row = mBase + e + hsel * 8;
      C[(size_t)row * NCOL + col] = acc[t][e] + bi;
    }
  }
}

// ---------------------------------------------------------------------------
// Kernel 2: per (n,g) row of 320 logits (one wave32 per row, 8 rows/block):
//  softmax(logits) -> marginal (LDS ds_add_f32 + one global atomic pass);
//  argmax(logits + gumbel) -> gather codevector row into out (float4).
// ---------------------------------------------------------------------------
__global__ __launch_bounds__(256)
void vq_rows_kernel(const float* __restrict__ logits, const float* __restrict__ gu,
                    const float* __restrict__ cv, float* __restrict__ out,
                    float* __restrict__ gmarg) {
  __shared__ float smarg[NCOL];
  const int tid = threadIdx.x;
  for (int i = tid; i < NCOL; i += 256) smarg[i] = 0.f;
  __syncthreads();

  const int lane = tid & 31;
  const int wave = tid >> 5;
  const int row  = blockIdx.x * 8 + wave;   // row in [0, NROW)
  const int n    = row >> 1;
  const int g    = row & 1;

  float l[10], z[10];
  float m = -3.4e38f;
#pragma unroll
  for (int j = 0; j < 10; ++j) {
    const int v = lane + 32 * j;
    l[j] = logits[(size_t)row * NV + v];
    const float u = gu[(size_t)row * NV + v];
    z[j] = l[j] - __logf(-__logf(u));      // logits + gumbel noise
    m = fmaxf(m, l[j]);
  }
#pragma unroll
  for (int o = 16; o; o >>= 1) m = fmaxf(m, __shfl_xor(m, o, 32));

  float e[10], s = 0.f;
#pragma unroll
  for (int j = 0; j < 10; ++j) { e[j] = __expf(l[j] - m); s += e[j]; }
#pragma unroll
  for (int o = 16; o; o >>= 1) s += __shfl_xor(s, o, 32);
  const float inv = 1.0f / s;
#pragma unroll
  for (int j = 0; j < 10; ++j)
    atomicAdd(&smarg[g * NV + lane + 32 * j], e[j] * inv);   // ds_add_f32

  float zb = -3.4e38f; int ib = 0;
#pragma unroll
  for (int j = 0; j < 10; ++j) {
    const int v = lane + 32 * j;
    if (z[j] > zb || (z[j] == zb && v < ib)) { zb = z[j]; ib = v; }
  }
#pragma unroll
  for (int o = 16; o; o >>= 1) {
    const float oz = __shfl_xor(zb, o, 32);
    const int   oi = __shfl_xor(ib, o, 32);
    if (oz > zb || (oz == zb && oi < ib)) { zb = oz; ib = oi; }
  }

  const float4* src = (const float4*)(cv + (size_t)(g * NV + ib) * CVD);
  float4* dst = (float4*)(out + (size_t)n * (NG * CVD) + g * CVD);
  dst[lane] = src[lane];

  __syncthreads();
  for (int i = tid; i < NCOL; i += 256) atomicAdd(&gmarg[i], smarg[i]);
}

// ---------------------------------------------------------------------------
// Kernel 3: perplexity = sum_g exp(-sum_v mbar*log(mbar+1e-7)), mbar = sum/N
// ---------------------------------------------------------------------------
__global__ void perplexity_kernel(const float* __restrict__ gmarg,
                                  float* __restrict__ out, int outIdx) {
  __shared__ float sg[2];
  if (threadIdx.x < 2) sg[threadIdx.x] = 0.f;
  __syncthreads();
  for (int i = threadIdx.x; i < NCOL; i += blockDim.x) {
    const float mb = gmarg[i] * (1.0f / (float)NTOK);
    atomicAdd(&sg[i / NV], mb * __logf(mb + 1e-7f));
  }
  __syncthreads();
  if (threadIdx.x == 0) out[outIdx] = __expf(-sg[0]) + __expf(-sg[1]);
}

__global__ void zero_marg_kernel(float* __restrict__ g) {
  const int i = blockIdx.x * blockDim.x + threadIdx.x;
  if (i < NCOL) g[i] = 0.f;
}

extern "C" void kernel_launch(void* const* d_in, const int* in_sizes, int n_in,
                              void* d_out, int out_size, void* d_ws, size_t ws_size,
                              hipStream_t stream) {
  const float* H    = (const float*)d_in[0];   // [8,2048,1024]
  const float* W    = (const float*)d_in[1];   // [640,1024]
  const float* bias = (const float*)d_in[2];   // [640]
  const float* cv   = (const float*)d_in[3];   // [1,640,128]
  const float* gu   = (const float*)d_in[4];   // [32768,320]
  float* out = (float*)d_out;                  // [8,2048,256] + perplexity scalar
  float* ws  = (float*)d_ws;

  float*  gmarg  = ws;                                   // 640 floats
  float*  logits = ws + 1024;                            // 16384*640 floats
  __bf16* whi    = (__bf16*)(ws + 1024 + (size_t)NTOK * NCOL);
  __bf16* wlo    = whi + (size_t)NCOL * HID;

  zero_marg_kernel<<<dim3(3), dim3(256), 0, stream>>>(gmarg);
  convert_w_kernel<<<dim3((NCOL * HID + 255) / 256), dim3(256), 0, stream>>>(W, whi, wlo);
  gemm_logits_kernel<<<dim3(NCOL / 128, NTOK / 64), dim3(256), 0, stream>>>(H, whi, wlo, bias, logits);
  vq_rows_kernel<<<dim3(NROW / 8), dim3(256), 0, stream>>>(logits, gu, cv, out, gmarg);
  perplexity_kernel<<<dim3(1), dim3(256), 0, stream>>>(gmarg, out, out_size - 1);
}
